// SingleAttentionHead_36283883717468
// MI455X (gfx1250) — matrix operand order repaired and checked by reading, not compiled, not run on hardware
//
#include <hip/hip_runtime.h>

#define SEQ 8192
#define EMB 512
#define ATT 64

typedef __attribute__((ext_vector_type(16))) _Float16 v16h;
typedef __attribute__((ext_vector_type(8)))  _Float16 v8h;
typedef __attribute__((ext_vector_type(8)))  float    v8f;

// (1/sqrt(ATT)) * log2(e): folded into Q during projection so phase 2 does
// p = exp2(score) directly.
#define CEXP (0.125f * 1.44269504088896340736f)

static __device__ __forceinline__ v8f wmma_f16(v16h a, v16h b, v8f c) {
  // D = A(16x32 f16) * B(32x16 f16) + C(16x16 f32)
  return __builtin_amdgcn_wmma_f32_16x16x32_f16(false, a, false, b, (short)0, c,
                                                false, false);
}

static __device__ __forceinline__ v16h cat8(v8h lo, v8h hi) {
  return __builtin_shufflevector(lo, hi, 0, 1, 2, 3, 4, 5, 6, 7, 8, 9, 10, 11,
                                 12, 13, 14, 15);
}

// ---------------------------------------------------------------------------
// Phase 1: Q/K/V projections.  grid = (SEQ/16, 3), block = 128 (4 waves).
// blockIdx.y selects matrix (0=Q,1=K,2=V); wave = 16-wide N tile of the 64 atts.
// Qh (pre-scaled by CEXP), Kh row-major f16 [SEQ][64]; V transposed Vt [64][SEQ].
// ---------------------------------------------------------------------------
__global__ __launch_bounds__(128, 1)
void qkv_proj_kernel(const float* __restrict__ input,
                     const float* __restrict__ Wq, const float* __restrict__ bq,
                     const float* __restrict__ Wk, const float* __restrict__ bk,
                     const float* __restrict__ Wv, const float* __restrict__ bv,
                     _Float16* __restrict__ Qh, _Float16* __restrict__ Kh,
                     _Float16* __restrict__ Vt) {
  const int lane  = threadIdx.x & 31;
  const int wv    = threadIdx.x >> 5;  // N tile 0..3
  const int grp   = lane >> 4;         // lane half-group
  const int ln    = lane & 15;
  const int mbase = blockIdx.x * 16;
  const int nbase = wv * 16;
  const int mat   = blockIdx.y;

  const float* W  = (mat == 0) ? Wq : (mat == 1) ? Wk : Wv;
  const float* bb = (mat == 0) ? bq : (mat == 1) ? bk : bv;

  const float* arow = input + (size_t)(mbase + ln) * EMB;  // A: M = lane row
  const float* wrow = W + (size_t)(nbase + ln) * EMB;      // B: N = lane col

  v8f acc = {};
#pragma unroll
  for (int ks = 0; ks < EMB; ks += 32) {
    // A operand 16x32: lane holds K runs {grp*8..+7} and {grp*8+16..+23}
    const int alo = ks + grp * 8;
    v8f fa0 = *(const v8f*)(arow + alo);
    v8f fa1 = *(const v8f*)(arow + alo + 16);
    v16h a  = cat8(__builtin_convertvector(fa0, v8h),
                   __builtin_convertvector(fa1, v8h));
    // B operand 32x16: B[k][n] = W[n][ks+k]; lane holds K runs {grp*8..+7} and {grp*8+16..+23}
    const float* wp = wrow + ks + grp * 8;
    v8f fb0 = *(const v8f*)(wp);
    v8f fb1 = *(const v8f*)(wp + 16);
    v16h b  = cat8(__builtin_convertvector(fb0, v8h),
                   __builtin_convertvector(fb1, v8h));
    acc = wmma_f16(a, b, acc);
  }

  const float bn = bb[nbase + ln];
  const float scale = (mat == 0) ? CEXP : 1.0f;  // fold softmax scale into Q

  // C layout: VGPR r -> row (r + 8*grp), lane%16 -> col
#pragma unroll
  for (int r = 0; r < 8; ++r) {
    const int m = mbase + r + 8 * grp;
    const _Float16 h = (_Float16)((acc[r] + bn) * scale);
    if (mat == 0)
      Qh[(size_t)m * ATT + nbase + ln] = h;
    else if (mat == 1)
      Kh[(size_t)m * ATT + nbase + ln] = h;
    else
      Vt[(size_t)(nbase + ln) * SEQ + m] = h;  // transposed for phase 2
  }
}

// ---------------------------------------------------------------------------
// Phase 2: flash attention, fully transposed, LDS-free.
//   S^T = K . Q^T           (key on VGPR axis, query on lane axis)
//   P^T = exp2(S^T)         (scale pre-folded into Qh; scores are O(1), fp32
//                            exp needs no max-stabilization; softmax exact)
//   O^T += V^T . P^T  (exp output packs straight into the B operand;
//                            V^T A operand uses the same key order:
//                            group g loads keys n0+g*8..+7 and
//                            n0+g*8+16..+23, as P^T packs them)
// Denominator per query is a per-lane scalar -> one xor-16 shuffle at the end.
// grid = SEQ/32, block = 64 (2 waves); one wave per 16-query block.
// ---------------------------------------------------------------------------
__global__ __launch_bounds__(64, 1)
void flash_attn_kernel(const _Float16* __restrict__ Qh,
                       const _Float16* __restrict__ Kh,
                       const _Float16* __restrict__ Vt,
                       float* __restrict__ out) {
  const int lane  = threadIdx.x & 31;
  const int wv    = threadIdx.x >> 5;
  const int grp   = lane >> 4;
  const int ln    = lane & 15;
  const int mbase = (blockIdx.x * 2 + wv) * 16;

  // Q^T as B operand: lane = query, halves = emb dims grp*8..+7 and grp*8+16..+23 (+32/step)
  const _Float16* qrow = Qh + (size_t)(mbase + ln) * ATT + grp * 8;
  v16h qb0 = cat8(*(const v8h*)(qrow), *(const v8h*)(qrow + 16));       // d 0..31
  v16h qb1 = cat8(*(const v8h*)(qrow + 32), *(const v8h*)(qrow + 48));  // d 32..63

  v8f o0 = {}, o1 = {}, o2 = {}, o3 = {};  // O^T tiles: att 16t + r + 8*grp, q = ln
  float psum = 0.0f;                       // per-lane partial softmax denom

  const _Float16* vt0 = Vt + (size_t)(0 + ln) * SEQ;
  const _Float16* vt1 = Vt + (size_t)(16 + ln) * SEQ;
  const _Float16* vt2 = Vt + (size_t)(32 + ln) * SEQ;
  const _Float16* vt3 = Vt + (size_t)(48 + ln) * SEQ;
  const int dlo = grp * 8;

  for (int n0 = 0; n0 < SEQ; n0 += 32) {
    // K rows as A operand: two 16-key tiles x two 32-d steps
    const _Float16* kr0 = Kh + (size_t)(n0 + ln) * ATT;
    const _Float16* kr1 = kr0 + 16 * ATT;
    v16h ka00 = cat8(*(const v8h*)(kr0 + dlo), *(const v8h*)(kr0 + dlo + 16));
    v16h ka01 = cat8(*(const v8h*)(kr0 + 32 + dlo),
                     *(const v8h*)(kr0 + 32 + dlo + 16));
    v16h ka10 = cat8(*(const v8h*)(kr1 + dlo), *(const v8h*)(kr1 + dlo + 16));
    v16h ka11 = cat8(*(const v8h*)(kr1 + 32 + dlo),
                     *(const v8h*)(kr1 + 32 + dlo + 16));

    // prefetch next key block (K rows and V columns) into WGP$
    if (n0 + 32 < SEQ) {
      __builtin_prefetch((const void*)(kr0 + 32 * ATT), 0, 3);
      __builtin_prefetch((const void*)(kr1 + 32 * ATT), 0, 3);
      __builtin_prefetch((const void*)(vt0 + n0 + 32 + grp * 16), 0, 3);
      __builtin_prefetch((const void*)(vt2 + n0 + 32 + grp * 16), 0, 3);
    }

    v8f s0 = {}, s1 = {};
    s0 = wmma_f16(ka00, qb0, s0);
    s0 = wmma_f16(ka01, qb1, s0);
    s1 = wmma_f16(ka10, qb0, s1);
    s1 = wmma_f16(ka11, qb1, s1);

    // P^T = exp2(S^T); packs directly into the B operand (key = VGPR axis)
    v16h pb;
#pragma unroll
    for (int r = 0; r < 8; ++r) {
      float p0 = __builtin_amdgcn_exp2f(s0[r]);
      float p1 = __builtin_amdgcn_exp2f(s1[r]);
      psum += p0 + p1;
      pb[r]     = (_Float16)p0;  // halves 0..7  <- key tile 0 (keys r+8g)
      pb[8 + r] = (_Float16)p1;  // halves 8..15 <- key tile 1 (keys 16+r+8g)
    }

    // V^T rows as A operand; key runs {grp*8..+7} and {grp*8+16..+23}
    v16h va0 = cat8(*(const v8h*)(vt0 + n0 + grp * 8), *(const v8h*)(vt0 + n0 + grp * 8 + 16));
    v16h va1 = cat8(*(const v8h*)(vt1 + n0 + grp * 8), *(const v8h*)(vt1 + n0 + grp * 8 + 16));
    v16h va2 = cat8(*(const v8h*)(vt2 + n0 + grp * 8), *(const v8h*)(vt2 + n0 + grp * 8 + 16));
    v16h va3 = cat8(*(const v8h*)(vt3 + n0 + grp * 8), *(const v8h*)(vt3 + n0 + grp * 8 + 16));
    o0 = wmma_f16(va0, pb, o0);
    o1 = wmma_f16(va1, pb, o1);
    o2 = wmma_f16(va2, pb, o2);
    o3 = wmma_f16(va3, pb, o3);
  }

  // complete the softmax denominator: other lane-group holds the other keys
  const float tot = psum + __shfl_xor(psum, 16, 32);
  const float inv = 1.0f / tot;

  // O^T store: per lane 8 contiguous atts per tile -> vectorized b128 stores
  float* orow = out + (size_t)(mbase + ln) * ATT + 8 * grp;
  v8f ov0, ov1, ov2, ov3;
#pragma unroll
  for (int r = 0; r < 8; ++r) {
    ov0[r] = o0[r] * inv;
    ov1[r] = o1[r] * inv;
    ov2[r] = o2[r] * inv;
    ov3[r] = o3[r] * inv;
  }
  *(v8f*)(orow + 0)  = ov0;
  *(v8f*)(orow + 16) = ov1;
  *(v8f*)(orow + 32) = ov2;
  *(v8f*)(orow + 48) = ov3;
}

// ---------------------------------------------------------------------------
extern "C" void kernel_launch(void* const* d_in, const int* in_sizes, int n_in,
                              void* d_out, int out_size, void* d_ws,
                              size_t ws_size, hipStream_t stream) {
  const float* input = (const float*)d_in[0];
  const float* Wq    = (const float*)d_in[1];
  const float* bq    = (const float*)d_in[2];
  const float* Wk    = (const float*)d_in[3];
  const float* bk    = (const float*)d_in[4];
  const float* Wv    = (const float*)d_in[5];
  const float* bv    = (const float*)d_in[6];
  float* out         = (float*)d_out;

  _Float16* Qh = (_Float16*)d_ws;         // SEQ*ATT f16  (1 MB, pre-scaled)
  _Float16* Kh = Qh + (size_t)SEQ * ATT;  // SEQ*ATT f16  (1 MB)
  _Float16* Vt = Kh + (size_t)SEQ * ATT;  // ATT*SEQ f16  (1 MB, transposed)

  dim3 g1(SEQ / 16, 3, 1);
  qkv_proj_kernel<<<g1, 128, 0, stream>>>(input, Wq, bq, Wk, bk, Wv, bv, Qh, Kh,
                                          Vt);

  dim3 g2(SEQ / 32, 1, 1);
  flash_attn_kernel<<<g2, 64, 0, stream>>>(Qh, Kh, Vt, out);
}
